// MultiHeadAttentionEuclid_39109972198043
// MI455X (gfx1250) — matrix operand order repaired and checked by reading, not compiled, not run on hardware
//
#include <hip/hip_runtime.h>

#define B_  4
#define T_  1024
#define D_  1024
#define H_  16
#define HD_ 64
#define SBLK 128   // K/V staging tile width (s dimension)

typedef __attribute__((ext_vector_type(16))) _Float16 v16h;
typedef __attribute__((ext_vector_type(8)))  _Float16 h8;
typedef __attribute__((ext_vector_type(8)))  float    v8f;
typedef __attribute__((ext_vector_type(4)))  float    f4;

// ---------- WMMA helper ----------
static __device__ inline v8f wmma16(v16h a, v16h b, v8f c) {
  // D = A(16x32 f16) * B(32x16 f16) + C(16x16 f32)
  return __builtin_amdgcn_wmma_f32_16x16x32_f16(
      /*neg_a=*/false, a, /*neg_b=*/false, b,
      /*c_mod=*/(short)0, c, /*reuse_a=*/false, /*reuse_b=*/false);
}

// ---------- A-fragment loader (row per lane; K split kb / kb+16) ----------
static __device__ inline v16h fragA_f16(const _Float16* p, int stride) {
  const int lane = threadIdx.x & 31;
  const int row  = lane & 15;
  const int kb   = (lane >> 4) << 3;
  const _Float16* q = p + (size_t)row * stride + kb;
  h8 lo = *(const h8*)(q);
  h8 hi = *(const h8*)(q + 16);
  v16h f;
#pragma unroll
  for (int i = 0; i < 8; ++i) { f[i] = lo[i]; f[8 + i] = hi[i]; }
  return f;
}

// ---------- B-fragment loader (column per lane; K split kb / kb+16) ----
// Source stored N-major (row n holds K contiguous), i.e. B^T row-major.
static __device__ inline v16h fragB_f16(const _Float16* p, int stride) {
  const int lane = threadIdx.x & 31;
  const int col  = lane & 15;
  const int k0   = (lane >> 4) << 3;
  const _Float16* q = p + (size_t)col * stride + k0;
  h8 lo = *(const h8*)(q);
  h8 hi = *(const h8*)(q + 16);
  v16h f;
#pragma unroll
  for (int i = 0; i < 8; ++i) { f[i] = lo[i]; f[8 + i] = hi[i]; }
  return f;
}

// Low 32 bits of a generic shared-memory pointer = LDS byte address
// (ISA 10.2: LDS aperture uses addr[31:0] as LDS_ADDR).
static __device__ inline unsigned lds_addr32(const void* p) {
  return (unsigned)(unsigned long long)p;
}

// Async copy 16B per lane: global -> LDS, tracked by ASYNCcnt.
static __device__ inline void async_b128(unsigned lds_byte, const void* gaddr) {
  asm volatile("global_load_async_to_lds_b128 %0, %1, off"
               :: "v"(lds_byte), "v"((unsigned long long)gaddr) : "memory");
}
static __device__ inline void wait_async0() {
  asm volatile("s_wait_asynccnt 0x0" ::: "memory");
}

// =====================================================================
// Kernel 0: streaming f32 -> f16 conversion (x and the four weights).
// =====================================================================
__global__ __launch_bounds__(256) void cvt_f32_to_f16(
    const float* __restrict__ src, _Float16* __restrict__ dst)
{
  const size_t i = ((size_t)blockIdx.x * blockDim.x + threadIdx.x) * 8;
  f4 a = *(const f4*)(src + i);
  f4 b = *(const f4*)(src + i + 4);
  h8 o;
#pragma unroll
  for (int j = 0; j < 4; ++j) {
    o[j]     = (_Float16)a[j];
    o[4 + j] = (_Float16)b[j];
  }
  *(h8*)(dst + i) = o;
}

// =====================================================================
// Kernel 1: fused QKV projection + bias + RoPE. One wave per 32x64
// output tile: 2 A-frags x 4 B-frags -> 8 WMMAs per 32-wide k-step.
// Q is pre-scaled by 1/sqrt(HD). Q/K stored f16 [B,H,T,HD]; V stored
// transposed [B,H,HD,T] so P*V B-fragments are contiguous.
// =====================================================================
__global__ __launch_bounds__(256) void mha_qkv_rope(
    const _Float16* __restrict__ xh,
    const _Float16* __restrict__ Wqh, const float* __restrict__ bq,
    const _Float16* __restrict__ Wkh, const float* __restrict__ bk,
    const _Float16* __restrict__ Wvh, const float* __restrict__ bv,
    _Float16* __restrict__ Qh, _Float16* __restrict__ Kh,
    _Float16* __restrict__ Vt)
{
  const int wid   = (blockIdx.x * blockDim.x + threadIdx.x) >> 5;
  const int TN    = 3 * D_ / 64;             // 48 column tiles (Q|K|V)
  const int tm    = wid / TN;                // 0..127 (B*T/32)
  const int tn    = wid % TN;
  const int which = tn >> 4;                 // 0=Q 1=K 2=V
  const int col0  = (tn & 15) << 6;

  const _Float16* W    = (which == 0) ? Wqh : (which == 1 ? Wkh : Wvh);
  const float*    bias = (which == 0) ? bq  : (which == 1 ? bk  : bv);

  const int lane = threadIdx.x & 31;
  const int n    = lane & 15;
  const int hf   = lane >> 4;

  const _Float16* xrow = xh + (size_t)(tm * 32) * D_;
  const _Float16* wrow = W + (size_t)col0 * D_;

  v8f acc[2][4] = {};
  for (int k = 0; k < D_; k += 32) {
    if (k + 64 < D_) {
      __builtin_prefetch(xrow + k + 64, 0, 3);
      __builtin_prefetch(wrow + k + 64, 0, 3);
    }
    v16h a0 = fragA_f16(xrow + k, D_);
    v16h a1 = fragA_f16(xrow + (size_t)16 * D_ + k, D_);
    v16h b0 = fragB_f16(wrow + k, D_);
    v16h b1 = fragB_f16(wrow + (size_t)16 * D_ + k, D_);
    v16h b2 = fragB_f16(wrow + (size_t)32 * D_ + k, D_);
    v16h b3 = fragB_f16(wrow + (size_t)48 * D_ + k, D_);
    acc[0][0] = wmma16(a0, b0, acc[0][0]);
    acc[0][1] = wmma16(a0, b1, acc[0][1]);
    acc[0][2] = wmma16(a0, b2, acc[0][2]);
    acc[0][3] = wmma16(a0, b3, acc[0][3]);
    acc[1][0] = wmma16(a1, b0, acc[1][0]);
    acc[1][1] = wmma16(a1, b1, acc[1][1]);
    acc[1][2] = wmma16(a1, b2, acc[1][2]);
    acc[1][3] = wmma16(a1, b3, acc[1][3]);
  }

  const bool doRope = (which < 2);
  const bool odd    = (n & 1);
  _Float16* dstQK   = (which == 0) ? Qh : Kh;
  const float qscale = (which == 0) ? 0.125f : 1.0f;   // fold 1/sqrt(HD) into Q

#pragma unroll
  for (int j = 0; j < 4; ++j) {
    const int   e    = col0 + j * 16 + n;    // output feature
    const float bval = bias[e];
    const int   i2   = (e & (HD_ - 1)) >> 1;
    const float inv_freq =
        __expf(-(float)i2 * (9.21034037197618f / 32.0f)); // 10000^(-i2/32)
    const int h = e >> 6;
    const int p = e & (HD_ - 1);
#pragma unroll
    for (int i = 0; i < 2; ++i) {
#pragma unroll
      for (int r = 0; r < 8; ++r) {
        float v = acc[i][j][r] + bval;
        const int tok = tm * 32 + i * 16 + r + hf * 8;
        const int t   = tok & (T_ - 1);
        if (doRope) {
          float sv, cv;
          __sincosf((float)t * inv_freq, &sv, &cv);
          const float other = __shfl_xor(v, 1, 32);
          v = odd ? (v * cv + other * sv) : (v * cv - other * sv);
        }
        const int bb = tok >> 10;            // batch
        const _Float16 val = (_Float16)(v * qscale);
        if (which < 2) {
          dstQK[(((size_t)(bb * H_ + h) * T_) + t) * HD_ + p] = val;
        } else {
          Vt[((size_t)(bb * H_ + h) * HD_ + p) * T_ + t] = val;
        }
      }
    }
  }
}

// =====================================================================
// Kernel 2: causal flash attention. 8 waves/block share one (b,h); K/V
// tiles (128 wide) are staged into LDS once per block with
// global_load_async_to_lds_b128 (ASYNCcnt), then every wave fragment-
// loads from LDS. One wave per 16-row q tile; online softmax in f32;
// P staged through LDS for the D-frag -> A-frag relayout.
// =====================================================================
__global__ __launch_bounds__(256) void mha_flash_attn(
    const _Float16* __restrict__ Qh, const _Float16* __restrict__ Kh,
    const _Float16* __restrict__ Vt, _Float16* __restrict__ Ctx)
{
  __shared__ alignas(16) _Float16 Ktile[SBLK][HD_];    // 16 KB
  __shared__ alignas(16) _Float16 Vtile[HD_][SBLK];    // 16 KB
  __shared__ alignas(16) _Float16 plds[8][16][32];     // 8 KB

  const int w    = threadIdx.x >> 5;
  const int wid  = blockIdx.x * 8 + w;
  const int qt   = wid & 63;                 // q tile within (b,h)
  const int bh   = wid >> 6;                 // same for all 8 waves of block
  const int q0   = qt << 4;
  const int lane = threadIdx.x & 31;
  const int n    = lane & 15;
  const int hf   = lane >> 4;
  const int tid  = threadIdx.x;

  // block-uniform causal bound: last wave's q0 + 16 (multiple of 128)
  const int smax = ((blockIdx.x * 8) & 63) * 16 + SBLK;

  const _Float16* qbase = Qh + ((size_t)bh * T_ + q0) * HD_;
  const _Float16* kbase = Kh + (size_t)bh * T_ * HD_;
  const _Float16* vbase = Vt + (size_t)bh * HD_ * T_;

  const v16h qf0 = fragA_f16(qbase, HD_);        // d = 0..31
  const v16h qf1 = fragA_f16(qbase + 32, HD_);   // d = 32..63

  float mrow[8], lrow[8];
  v8f o0 = {}, o1 = {}, o2 = {}, o3 = {};
#pragma unroll
  for (int r = 0; r < 8; ++r) { mrow[r] = -1e30f; lrow[r] = 0.0f; }

  const unsigned kt0 = lds_addr32(&Ktile[0][0]);
  const unsigned vt0 = lds_addr32(&Vtile[0][0]);

  for (int s0 = 0; s0 < smax; s0 += SBLK) {
    __syncthreads();   // previous tile fully consumed by all waves

    // ---- async stage K tile: 128 rows x 64 halfs, contiguous 16 KB ----
    {
      const char* gk = (const char*)(kbase + (size_t)s0 * HD_);
#pragma unroll
      for (int j = 0; j < 4; ++j) {
        const unsigned off = (unsigned)(tid + j * 256) * 16;
        async_b128(kt0 + off, gk + off);
      }
    }
    // ---- async stage V tile: 64 rows (stride T) x 128 halfs ----
    {
#pragma unroll
      for (int j = 0; j < 4; ++j) {
        const int c   = tid + j * 256;       // 1024 16B-chunks
        const int row = c >> 4;
        const int cp  = c & 15;
        const char* gv =
            (const char*)(vbase + (size_t)row * T_ + s0) + cp * 16;
        async_b128(vt0 + (unsigned)(row * (SBLK * 2) + cp * 16), gv);
      }
    }
    wait_async0();
    __syncthreads();   // tile visible to all waves

    // ---- compute on the staged tile (wave-uniform causal skip) ----
    for (int ss = 0; ss < SBLK; ss += 32) {
      const int sb0 = s0 + ss;
      if (sb0 >= q0 + 16) break;             // wave-uniform

      const _Float16* krow = &Ktile[ss][0];
      v8f S0 = {}, S1 = {};
      S0 = wmma16(qf0, fragB_f16(krow, HD_), S0);
      S0 = wmma16(qf1, fragB_f16(krow + 32, HD_), S0);
      S1 = wmma16(qf0, fragB_f16(krow + 16 * HD_, HD_), S1);
      S1 = wmma16(qf1, fragB_f16(krow + 16 * HD_ + 32, HD_), S1);

      float alpha[8];
#pragma unroll
      for (int r = 0; r < 8; ++r) {
        const int t  = q0 + r + hf * 8;
        const int sa = sb0 + n;
        const int sb = sb0 + 16 + n;
        const float v0 = (sa <= t) ? S0[r] : -3.0e38f; // scale folded into Q
        const float v1 = (sb <= t) ? S1[r] : -3.0e38f;

        float rmax = fmaxf(v0, v1);
#pragma unroll
        for (int off = 1; off < 16; off <<= 1)
          rmax = fmaxf(rmax, __shfl_xor(rmax, off, 32));

        const float mnew = fmaxf(mrow[r], rmax);
        const float a    = __expf(mrow[r] - mnew);
        const float p0   = __expf(v0 - mnew);
        const float p1   = __expf(v1 - mnew);

        float rs = p0 + p1;
#pragma unroll
        for (int off = 1; off < 16; off <<= 1)
          rs += __shfl_xor(rs, off, 32);

        lrow[r]  = lrow[r] * a + rs;
        mrow[r]  = mnew;
        alpha[r] = a;

        const int m = r + hf * 8;
        plds[w][m][n]      = (_Float16)p0;   // D-frag -> [m][s] layout
        plds[w][m][n + 16] = (_Float16)p1;
      }

#pragma unroll
      for (int r = 0; r < 8; ++r) {
        o0[r] *= alpha[r]; o1[r] *= alpha[r];
        o2[r] *= alpha[r]; o3[r] *= alpha[r];
      }

      // Re-read P as an A-fragment (16x32 over s)
      const int akb = hf << 3;
      h8 lo = *(const h8*)&plds[w][n][akb];
      h8 hi = *(const h8*)&plds[w][n][akb + 16];
      v16h pf;
#pragma unroll
      for (int i = 0; i < 8; ++i) { pf[i] = lo[i]; pf[8 + i] = hi[i]; }

      const _Float16* vrow = &Vtile[0][ss];
      o0 = wmma16(pf, fragB_f16(vrow + (size_t)(0 * 16) * SBLK, SBLK), o0);
      o1 = wmma16(pf, fragB_f16(vrow + (size_t)(1 * 16) * SBLK, SBLK), o1);
      o2 = wmma16(pf, fragB_f16(vrow + (size_t)(2 * 16) * SBLK, SBLK), o2);
      o3 = wmma16(pf, fragB_f16(vrow + (size_t)(3 * 16) * SBLK, SBLK), o3);
    }
  }

  const int b = bh >> 4;
  const int h = bh & 15;
#pragma unroll
  for (int r = 0; r < 8; ++r) {
    const int tok = b * T_ + q0 + r + hf * 8;
    const float inv = 1.0f / lrow[r];
    _Float16* dst = Ctx + (size_t)tok * D_ + h * HD_;
    dst[0 * 16 + n] = (_Float16)(o0[r] * inv);
    dst[1 * 16 + n] = (_Float16)(o1[r] * inv);
    dst[2 * 16 + n] = (_Float16)(o2[r] * inv);
    dst[3 * 16 + n] = (_Float16)(o3[r] * inv);
  }
}

// =====================================================================
// Kernel 3: output projection ctx @ Wo^T + bo -> f32. 32x64 per wave.
// =====================================================================
__global__ __launch_bounds__(256) void mha_oproj(
    const _Float16* __restrict__ Ctx, const _Float16* __restrict__ Woh,
    const float* __restrict__ bo, float* __restrict__ out)
{
  const int wid  = (blockIdx.x * blockDim.x + threadIdx.x) >> 5;
  const int TN   = D_ / 64;                  // 16
  const int tm   = wid / TN;                 // 0..127
  const int col0 = (wid % TN) << 6;
  const int lane = threadIdx.x & 31;
  const int n    = lane & 15;
  const int hf   = lane >> 4;

  const _Float16* crow = Ctx + (size_t)(tm * 32) * D_;
  const _Float16* wrow = Woh + (size_t)col0 * D_;

  v8f acc[2][4] = {};
  for (int k = 0; k < D_; k += 32) {
    if (k + 64 < D_) {
      __builtin_prefetch(crow + k + 64, 0, 3);
      __builtin_prefetch(wrow + k + 64, 0, 3);
    }
    v16h a0 = fragA_f16(crow + k, D_);
    v16h a1 = fragA_f16(crow + (size_t)16 * D_ + k, D_);
    v16h b0 = fragB_f16(wrow + k, D_);
    v16h b1 = fragB_f16(wrow + (size_t)16 * D_ + k, D_);
    v16h b2 = fragB_f16(wrow + (size_t)32 * D_ + k, D_);
    v16h b3 = fragB_f16(wrow + (size_t)48 * D_ + k, D_);
    acc[0][0] = wmma16(a0, b0, acc[0][0]);
    acc[0][1] = wmma16(a0, b1, acc[0][1]);
    acc[0][2] = wmma16(a0, b2, acc[0][2]);
    acc[0][3] = wmma16(a0, b3, acc[0][3]);
    acc[1][0] = wmma16(a1, b0, acc[1][0]);
    acc[1][1] = wmma16(a1, b1, acc[1][1]);
    acc[1][2] = wmma16(a1, b2, acc[1][2]);
    acc[1][3] = wmma16(a1, b3, acc[1][3]);
  }

#pragma unroll
  for (int j = 0; j < 4; ++j) {
    const float bval = bo[col0 + j * 16 + n];
#pragma unroll
    for (int i = 0; i < 2; ++i) {
#pragma unroll
      for (int r = 0; r < 8; ++r) {
        const int m = i * 16 + r + hf * 8;
        out[(size_t)(tm * 32 + m) * D_ + col0 + j * 16 + n] =
            acc[i][j][r] + bval;
      }
    }
  }
}

// =====================================================================
extern "C" void kernel_launch(void* const* d_in, const int* in_sizes, int n_in,
                              void* d_out, int out_size, void* d_ws, size_t ws_size,
                              hipStream_t stream) {
  const float* x  = (const float*)d_in[0];
  const float* Wq = (const float*)d_in[1];
  const float* bq = (const float*)d_in[2];
  const float* Wk = (const float*)d_in[3];
  const float* bk = (const float*)d_in[4];
  const float* Wv = (const float*)d_in[5];
  const float* bv = (const float*)d_in[6];
  const float* Wo = (const float*)d_in[7];
  const float* bo = (const float*)d_in[8];
  // d_in[9] = causal mask (tril) -- causality is hardcoded in the kernels.
  float* out = (float*)d_out;

  const size_t NX  = (size_t)B_ * T_ * D_;       // 4M elements
  const size_t NW  = (size_t)D_ * D_;            // 1M elements
  const size_t NQK = (size_t)B_ * H_ * T_ * HD_; // 4M elements

  _Float16* xh  = (_Float16*)d_ws;
  _Float16* Wqh = xh + NX;
  _Float16* Wkh = Wqh + NW;
  _Float16* Wvh = Wkh + NW;
  _Float16* Woh = Wvh + NW;
  _Float16* Qh  = Woh + NW;
  _Float16* Kh  = Qh + NQK;
  _Float16* Vt  = Kh + NQK;
  _Float16* Ctx = Vt + NQK;

  // f32 -> f16 pre-conversion (2048 elements per 256-thread block)
  cvt_f32_to_f16<<<(int)(NX / 2048), 256, 0, stream>>>(x, xh);
  cvt_f32_to_f16<<<(int)(NW / 2048), 256, 0, stream>>>(Wq, Wqh);
  cvt_f32_to_f16<<<(int)(NW / 2048), 256, 0, stream>>>(Wk, Wkh);
  cvt_f32_to_f16<<<(int)(NW / 2048), 256, 0, stream>>>(Wv, Wvh);
  cvt_f32_to_f16<<<(int)(NW / 2048), 256, 0, stream>>>(Wo, Woh);

  // QKV+RoPE: (B*T/32)*(3*D/64) = 6144 wave tiles / 8 waves per block
  mha_qkv_rope<<<768, 256, 0, stream>>>(xh, Wqh, bq, Wkh, bk, Wvh, bv,
                                        Qh, Kh, Vt);
  // Attention: B*H*(T/16) = 4096 waves, 8 waves/block sharing (b,h)
  mha_flash_attn<<<512, 256, 0, stream>>>(Qh, Kh, Vt, Ctx);
  // O-proj: (B*T/32)*(D/64) = 2048 waves
  mha_oproj<<<256, 256, 0, stream>>>(Ctx, Woh, bo, out);
}
